// MultiheadAttention_26113401159824
// MI455X (gfx1250) — matrix operand order repaired and checked by reading, not compiled, not run on hardware
//
#include <hip/hip_runtime.h>

// ---------------------------------------------------------------------------
// CDNA5 (gfx1250) wave32 WMMA multihead-attention forward.
// B=2, S=2048, D_IN=D_EMB=1024, H=16, HD=64.
// GEMMs: B operand staged via global_load_async_to_lds_b128 (ASYNCcnt,
// double-buffered LDS), A operand register double-buffered.
// Attention: flash-style, wave-private P tile, s_wait_dscnt wave-local sync.
// ---------------------------------------------------------------------------

typedef __attribute__((ext_vector_type(16))) __bf16 v16bf;
typedef __attribute__((ext_vector_type(8)))  __bf16 v8bf;
typedef __attribute__((ext_vector_type(8)))  float  v8f;

#define WMMA_BF16(a, b, c) \
  __builtin_amdgcn_wmma_f32_16x16x32_bf16(false, (a), false, (b), (short)0, (c), false, false)

// Async DMA: 16 bytes/lane global -> LDS, tracked by ASYNCcnt.
static __device__ __forceinline__ void async_copy_b128(unsigned lds_off, const void* gaddr) {
  asm volatile("global_load_async_to_lds_b128 %0, %1, off"
               :: "v"(lds_off), "v"(gaddr) : "memory");
}
static __device__ __forceinline__ void wait_async(int n) {
  if (n == 0) asm volatile("s_wait_asynccnt 0x0" ::: "memory");
  else        asm volatile("s_wait_asynccnt 0x1" ::: "memory");
}
static __device__ __forceinline__ void wait_lds() {
  asm volatile("s_wait_dscnt 0x0" ::: "memory");
}

// A-fragment (16xK row-major tile): lane L -> row L&15.
// lanes<16 hold K {0..7,16..23}; lanes>=16 hold K {8..15,24..31}  (ISA 7.12.2)
static __device__ __forceinline__ v16bf load_a_frag(const __bf16* tile, int ld, int lane) {
  const __bf16* p = tile + (lane & 15) * ld + ((lane & 16) ? 8 : 0);
  union { v16bf f; v8bf h[2]; } u;
  u.h[0] = *(const v8bf*)(p);
  u.h[1] = *(const v8bf*)(p + 16);
  return u.f;
}

// B-fragment: column n of B == row n of the row-major source.
// lanes<16: K {0..7,16..23}; lanes>=16: K {8..15,24..31} (matches A-fragment).
static __device__ __forceinline__ v16bf load_b_frag(const __bf16* tile, int ld, int lane) {
  const __bf16* p = tile + (lane & 15) * ld + ((lane & 16) ? 8 : 0);
  union { v16bf f; v8bf h[2]; } u;
  u.h[0] = *(const v8bf*)(p);
  u.h[1] = *(const v8bf*)(p + 16);
  return u.f;
}

// ---------------------------------------------------------------------------
__global__ void cvt_f32_bf16(const float* __restrict__ src, __bf16* __restrict__ dst, int n) {
  int i = blockIdx.x * blockDim.x + threadIdx.x;
  int stride = gridDim.x * blockDim.x;
  for (; i < n; i += stride) dst[i] = (__bf16)src[i];
}

// ---------------------------------------------------------------------------
// Shared GEMM skeleton: block = 256 threads = 8 waves, block tile 128M x 64N,
// wave tile 16x64, K accumulated in steps of 32.  B k-slice (64x32 bf16 = 4KB)
// async-DMA'd to double-buffered LDS, shared by all 8 waves.
// ---------------------------------------------------------------------------
#define GEMM_PROLOGUE(Xp, Wp)                                                  \
  const int lane  = threadIdx.x & 31;                                          \
  const int wave  = threadIdx.x >> 5;                                          \
  const int mbase = blockIdx.y * 128 + wave * 16;                              \
  const int nbase = blockIdx.x * 64;                                           \
  __shared__ __align__(16) __bf16 Bs[2][64][32];                               \
  const __bf16* Arow = (Xp) + (size_t)mbase * 1024;                            \
  const int srow = threadIdx.x >> 2;                                           \
  const int schk = (threadIdx.x & 3) * 8;                                      \
  const __bf16* Wst = (Wp) + (size_t)(nbase + srow) * 1024 + schk;             \
  async_copy_b128((unsigned)(size_t)&Bs[0][srow][schk], Wst);                  \
  v16bf a_cur = load_a_frag(Arow, 1024, lane);                                 \
  v8f acc[4] = {};                                                             \
  int buf = 0;                                                                 \
  for (int k0 = 0; k0 < 1024; k0 += 32) {                                      \
    const bool more = (k0 + 32) < 1024;                                        \
    const int  kn   = more ? (k0 + 32) : 0;                                    \
    __syncthreads(); /* all waves done reading buffer we now overwrite */      \
    if (more) {                                                                \
      async_copy_b128((unsigned)(size_t)&Bs[buf ^ 1][srow][schk], Wst + kn);   \
      wait_async(1); /* current slice done, next stays in flight */            \
    } else {                                                                   \
      wait_async(0);                                                           \
    }                                                                          \
    v16bf a_nxt = load_a_frag(Arow + kn, 1024, lane);                          \
    __syncthreads(); /* whole B k-slice visible to all waves */                \
    const __bf16* Bt = &Bs[buf][0][0];                                         \
    _Pragma("unroll")                                                          \
    for (int nt = 0; nt < 4; ++nt) {                                           \
      v16bf b = load_b_frag(Bt + nt * 16 * 32, 32, lane);                      \
      acc[nt] = WMMA_BF16(a_cur, b, acc[nt]);                                  \
    }                                                                          \
    a_cur = a_nxt;                                                             \
    buf ^= 1;                                                                  \
  }                                                                            \
  const int rr = (lane & 16) ? 8 : 0;                                          \
  const int nc = lane & 15;

// QKV projection: [4096,1024] x [3072,1024]^T -> scatter to Q/K/Vt (bf16).
__global__ __launch_bounds__(256) void qkv_gemm_wmma(
    const __bf16* __restrict__ X, const __bf16* __restrict__ W,
    const float* __restrict__ bias,
    __bf16* __restrict__ Qo, __bf16* __restrict__ Ko, __bf16* __restrict__ Vt) {
  GEMM_PROLOGUE(X, W)
#pragma unroll
  for (int nt = 0; nt < 4; ++nt) {
#pragma unroll
    for (int e = 0; e < 8; ++e) {
      int m = mbase + e + rr;               // global row = b*S + s
      int n = nbase + nt * 16 + nc;         // column in [0,3072)
      float v = acc[nt][e] + bias[n];
      int bb = m >> 11;                     // S = 2048
      int s  = m & 2047;
      int h  = n / 192;
      int j  = n - h * 192;
      if (j < 64) {                         // Q[b,h,s,j], fold 1/sqrt(HD)
        Qo[(((size_t)(bb * 16 + h) * 2048) + s) * 64 + j] = (__bf16)(v * 0.125f);
      } else if (j < 128) {                 // K[b,h,s,j-64]
        Ko[(((size_t)(bb * 16 + h) * 2048) + s) * 64 + (j - 64)] = (__bf16)v;
      } else {                              // V stored transposed: Vt[b,h,d,s]
        Vt[((size_t)(bb * 16 + h) * 64 + (j - 128)) * 2048 + s] = (__bf16)v;
      }
    }
  }
}

// Output projection: [4096,1024] x [1024,1024]^T + bias -> f32 d_out.
__global__ __launch_bounds__(256) void out_gemm_wmma(
    const __bf16* __restrict__ A, const __bf16* __restrict__ W,
    const float* __restrict__ bias, float* __restrict__ Out) {
  GEMM_PROLOGUE(A, W)
#pragma unroll
  for (int nt = 0; nt < 4; ++nt) {
#pragma unroll
    for (int e = 0; e < 8; ++e) {
      int m = mbase + e + rr;
      int n = nbase + nt * 16 + nc;
      Out[(size_t)m * 1024 + n] = acc[nt][e] + bias[n];
    }
  }
}

// ---------------------------------------------------------------------------
// Flash attention: block = 128 threads = 4 decoupled waves; wave owns a
// 16-row Q tile.  grid = (S/64, B*H).  Per 32-key chunk: 4 WMMAs (QK^T) +
// 4 WMMAs (PV).  Next K-fragments + current V-fragments are issued before the
// softmax VALU block so global latency hides under VALU + XDL co-execution.
// ---------------------------------------------------------------------------
__global__ __launch_bounds__(128) void flash_attn_wmma(
    const __bf16* __restrict__ Q, const __bf16* __restrict__ K,
    const __bf16* __restrict__ Vt, __bf16* __restrict__ Oo) {
  const int lane  = threadIdx.x & 31;
  const int wave  = threadIdx.x >> 5;
  const int bh    = blockIdx.y;                 // b*16 + h
  const int qbase = blockIdx.x * 64 + wave * 16;

  const __bf16* Qh = Q  + (size_t)bh * 2048 * 64;
  const __bf16* Kh = K  + (size_t)bh * 2048 * 64;
  const __bf16* Vh = Vt + (size_t)bh * 64 * 2048;

  __shared__ __align__(32) __bf16 Pl[4 * 16 * 32];
  __bf16* Pw = Pl + wave * (16 * 32);           // wave-private P tile

  v16bf qa0 = load_a_frag(Qh + (size_t)qbase * 64,      64, lane);
  v16bf qa1 = load_a_frag(Qh + (size_t)qbase * 64 + 32, 64, lane);

  float mrow[8], lrow[8];
#pragma unroll
  for (int e = 0; e < 8; ++e) { mrow[e] = -1e30f; lrow[e] = 0.0f; }
  v8f o0 = {}, o1 = {}, o2 = {}, o3 = {};

  const int rr = (lane & 16) ? 8 : 0;
  const int nc = lane & 15;

  // preload first K chunk fragments
  v16bf kb0 = load_b_frag(Kh, 64, lane);
  v16bf kb1 = load_b_frag(Kh + 32, 64, lane);
  v16bf kb2 = load_b_frag(Kh + (size_t)16 * 64, 64, lane);
  v16bf kb3 = load_b_frag(Kh + (size_t)16 * 64 + 32, 64, lane);

  for (int kt = 0; kt < 2048; kt += 32) {
    // ---- scores S = Q * K^T for keys [kt, kt+32)
    v8f s0 = {}, s1 = {};
    s0 = WMMA_BF16(qa0, kb0, s0);
    s0 = WMMA_BF16(qa1, kb1, s0);
    s1 = WMMA_BF16(qa0, kb2, s1);
    s1 = WMMA_BF16(qa1, kb3, s1);

    // ---- issue next-chunk K loads and current-chunk V loads early
    const int kn = (kt + 32 < 2048) ? kt + 32 : 0;
    v16bf kn0 = load_b_frag(Kh + (size_t)kn * 64, 64, lane);
    v16bf kn1 = load_b_frag(Kh + (size_t)kn * 64 + 32, 64, lane);
    v16bf kn2 = load_b_frag(Kh + (size_t)(kn + 16) * 64, 64, lane);
    v16bf kn3 = load_b_frag(Kh + (size_t)(kn + 16) * 64 + 32, 64, lane);
    v16bf vb0 = load_b_frag(Vh + (size_t)0 * 16 * 2048 + kt, 2048, lane);
    v16bf vb1 = load_b_frag(Vh + (size_t)1 * 16 * 2048 + kt, 2048, lane);
    v16bf vb2 = load_b_frag(Vh + (size_t)2 * 16 * 2048 + kt, 2048, lane);
    v16bf vb3 = load_b_frag(Vh + (size_t)3 * 16 * 2048 + kt, 2048, lane);

    // ---- online softmax; row reductions stay inside 16-lane half-groups,
    // matching the C/D layout (lanes 0-15: rows 0-7; lanes 16-31: rows 8-15)
#pragma unroll
    for (int e = 0; e < 8; ++e) {
      float a = s0[e], b = s1[e];
      float mx = fmaxf(a, b);
      mx = fmaxf(mx, __shfl_xor(mx, 1, 32));
      mx = fmaxf(mx, __shfl_xor(mx, 2, 32));
      mx = fmaxf(mx, __shfl_xor(mx, 4, 32));
      mx = fmaxf(mx, __shfl_xor(mx, 8, 32));
      float mnew = fmaxf(mrow[e], mx);
      float corr = __expf(mrow[e] - mnew);
      float p0 = __expf(a - mnew);
      float p1 = __expf(b - mnew);
      float sum = p0 + p1;
      sum += __shfl_xor(sum, 1, 32);
      sum += __shfl_xor(sum, 2, 32);
      sum += __shfl_xor(sum, 4, 32);
      sum += __shfl_xor(sum, 8, 32);
      lrow[e] = lrow[e] * corr + sum;
      mrow[e] = mnew;
      o0[e] *= corr; o1[e] *= corr; o2[e] *= corr; o3[e] *= corr;
      // stage P row-major [16 x 32] bf16 so it reloads as an A-fragment
      Pw[(e + rr) * 32 + nc]      = (__bf16)p0;
      Pw[(e + rr) * 32 + nc + 16] = (__bf16)p1;
    }
    // wave-local store->load visibility; no block barrier needed (P private)
    wait_lds();
    v16bf pa = load_a_frag(Pw, 32, lane);

    // ---- O += P * V  (B operand = rows of transposed V -> contiguous)
    o0 = WMMA_BF16(pa, vb0, o0);
    o1 = WMMA_BF16(pa, vb1, o1);
    o2 = WMMA_BF16(pa, vb2, o2);
    o3 = WMMA_BF16(pa, vb3, o3);

    kb0 = kn0; kb1 = kn1; kb2 = kn2; kb3 = kn3;
  }

  // ---- normalize and store attn output in [B*S, 1024] (head-major columns)
  const int b = bh >> 4, h = bh & 15;
#pragma unroll
  for (int e = 0; e < 8; ++e) {
    float inv = 1.0f / lrow[e];
    int s = qbase + e + rr;
    size_t row = ((size_t)b * 2048 + s) * 1024 + (size_t)h * 64 + nc;
    Oo[row + 0]  = (__bf16)(o0[e] * inv);
    Oo[row + 16] = (__bf16)(o1[e] * inv);
    Oo[row + 32] = (__bf16)(o2[e] * inv);
    Oo[row + 48] = (__bf16)(o3[e] * inv);
  }
}

// ---------------------------------------------------------------------------
extern "C" void kernel_launch(void* const* d_in, const int* in_sizes, int n_in,
                              void* d_out, int out_size, void* d_ws, size_t ws_size,
                              hipStream_t stream) {
  const float* x     = (const float*)d_in[0];   // [2,2048,1024]
  const float* W_qkv = (const float*)d_in[1];   // [3072,1024]
  const float* b_qkv = (const float*)d_in[2];   // [3072]
  const float* W_o   = (const float*)d_in[3];   // [1024,1024]
  const float* b_o   = (const float*)d_in[4];   // [1024]
  float* out = (float*)d_out;

  char* ws = (char*)d_ws;
  size_t off = 0;
  auto carve = [&](size_t bytes) {
    char* p = ws + off;
    off += (bytes + 255) & ~(size_t)255;
    return p;
  };
  __bf16* xb   = (__bf16*)carve((size_t)4096 * 1024 * 2);       // x bf16
  __bf16* wqb  = (__bf16*)carve((size_t)3072 * 1024 * 2);       // W_qkv bf16
  __bf16* wob  = (__bf16*)carve((size_t)1024 * 1024 * 2);       // W_o bf16
  __bf16* Qb   = (__bf16*)carve((size_t)32 * 2048 * 64 * 2);    // Q [b,h,s,d]
  __bf16* Kb   = (__bf16*)carve((size_t)32 * 2048 * 64 * 2);    // K [b,h,s,d]
  __bf16* Vtb  = (__bf16*)carve((size_t)32 * 64 * 2048 * 2);    // V [b,h,d,s]
  __bf16* attn = (__bf16*)carve((size_t)4096 * 1024 * 2);       // attn out
  (void)ws_size; (void)in_sizes; (void)n_in; (void)out_size;

  cvt_f32_bf16<<<dim3(1024), dim3(256), 0, stream>>>(x, xb, 4096 * 1024);
  cvt_f32_bf16<<<dim3(1024), dim3(256), 0, stream>>>(W_qkv, wqb, 3072 * 1024);
  cvt_f32_bf16<<<dim3(512),  dim3(256), 0, stream>>>(W_o, wob, 1024 * 1024);

  qkv_gemm_wmma<<<dim3(48, 32), dim3(256), 0, stream>>>(xb, wqb, b_qkv, Qb, Kb, Vtb);
  flash_attn_wmma<<<dim3(32, 32), dim3(128), 0, stream>>>(Qb, Kb, Vtb, attn);
  out_gemm_wmma<<<dim3(16, 32), dim3(256), 0, stream>>>(attn, wob, b_o, out);
}
